// EncoderDecoderAttention_61710090109670
// MI455X (gfx1250) — matrix operand order repaired and checked by reading, not compiled, not run on hardware
//
#include <hip/hip_runtime.h>
#include <hip/hip_bf16.h>
#include <math.h>

// ---------------------------------------------------------------------------
// EncoderDecoderAttention for MI455X (gfx1250, wave32, WMMA bf16)
// B=8, S=1024, D=1024, H=16, DH=64
// ---------------------------------------------------------------------------

typedef __bf16 bf16_t;
typedef __bf16 v16bf __attribute__((ext_vector_type(16)));
typedef __bf16 v8bf  __attribute__((ext_vector_type(8)));
typedef __bf16 v4bf  __attribute__((ext_vector_type(4)));
typedef float  v8f   __attribute__((ext_vector_type(8)));
typedef float  v4f   __attribute__((ext_vector_type(4)));
typedef int    v4i   __attribute__((ext_vector_type(4)));

#define BB 8
#define SS 1024
#define DD 1024
#define HH 16
#define DHH 64

// gfx1250 async global->LDS copy path (ASYNCcnt), with sync fallback
#if defined(__gfx1250__) && \
    __has_builtin(__builtin_amdgcn_global_load_async_to_lds_b128) && \
    __has_builtin(__builtin_amdgcn_s_wait_asynccnt)
#define HAS_ASYNC_LDS 1
#else
#define HAS_ASYNC_LDS 0
#endif

#define AS1 __attribute__((address_space(1)))
#define AS3 __attribute__((address_space(3)))

__device__ __forceinline__ v8f splat8(float v) {
    v8f r;
    for (int i = 0; i < 8; ++i) r[i] = v;
    return r;
}

__device__ __forceinline__ v16bf cat8(v8bf lo, v8bf hi) {
    return __builtin_shufflevector(lo, hi, 0,1,2,3,4,5,6,7,8,9,10,11,12,13,14,15);
}

__device__ __forceinline__ v8f wmma_bf16(v16bf a, v16bf b, v8f c) {
    // (neg_a, A, neg_b, B, c_mod, C, reuse_a, reuse_b)
    return __builtin_amdgcn_wmma_f32_16x16x32_bf16(false, a, false, b,
                                                   (short)0, c, false, false);
}

// ---------------------------------------------------------------------------
// Kernel 1: C[m,n] = (X[m,:] . W[n,:] + bias[n]) * scale, scattered bf16 into
// [B,H,S,DH] layout. kvMode==1 splits N=2048 into K (n<1024) and V (n>=1024).
// Block tile: 128(M) x 64(N), 256 threads = 8 waves, each wave 32x32.
// ---------------------------------------------------------------------------
__global__ __launch_bounds__(256)
void proj_gemm_kernel(const float* __restrict__ X,     // [8192,1024]
                      const float* __restrict__ W,     // [N,1024]
                      const float* __restrict__ bias,  // [N]
                      bf16_t* __restrict__ out0,       // Q or K  [B,H,S,DH]
                      bf16_t* __restrict__ out1,       // V (kvMode)
                      int kvMode, float scale)
{
    __shared__ bf16_t Asm[128 * 32];   // [m][k]
    __shared__ bf16_t Bsm[64 * 32];    // [n][k]  (row of W = output column)

    const int tid   = threadIdx.x;
    const int lane  = tid & 31;
    const int wave  = tid >> 5;
    const int half  = lane >> 4;
    const int l15   = lane & 15;
    const int waveM = wave >> 1;       // 0..3
    const int waveN = wave & 1;        // 0..1
    const int mBase = blockIdx.x * 128;
    const int nBase = blockIdx.y * 64;

    v8f acc[2][2];
    for (int mt = 0; mt < 2; ++mt)
        for (int nt = 0; nt < 2; ++nt) {
            int n = nBase + waveN * 32 + nt * 16 + l15;
            acc[mt][nt] = splat8(bias[n]);
        }

    for (int kk = 0; kk < DD; kk += 32) {
        // A tile: 128x32 f32 -> bf16 LDS (4096 elems, 16 per thread)
        for (int i = 0; i < 4; ++i) {
            int c    = tid + i * 256;          // 0..1023
            int row  = c >> 3;                 // 0..127
            int col4 = (c & 7) * 4;            // 0..28
            const float* src = X + (size_t)(mBase + row) * DD + kk + col4;
            if (kk + 32 < DD) __builtin_prefetch(src + 32, 0, 1);
            v4f f = *(const v4f*)src;
            v4bf h;
            h[0] = (bf16_t)f[0]; h[1] = (bf16_t)f[1];
            h[2] = (bf16_t)f[2]; h[3] = (bf16_t)f[3];
            *(v4bf*)(Asm + row * 32 + col4) = h;
        }
        // B tile: 64x32 f32 -> bf16 LDS (2048 elems, 8 per thread)
        for (int i = 0; i < 2; ++i) {
            int c    = tid + i * 256;          // 0..511
            int row  = c >> 3;                 // 0..63
            int col4 = (c & 7) * 4;
            const float* src = W + (size_t)(nBase + row) * DD + kk + col4;
            if (kk + 32 < DD) __builtin_prefetch(src + 32, 0, 1);
            v4f f = *(const v4f*)src;
            v4bf h;
            h[0] = (bf16_t)f[0]; h[1] = (bf16_t)f[1];
            h[2] = (bf16_t)f[2]; h[3] = (bf16_t)f[3];
            *(v4bf*)(Bsm + row * 32 + col4) = h;
        }
        __syncthreads();

        // A fragment: lane row m, K = {ko..ko+7, 16+ko..16+ko+7}, ko = half*8
        v16bf afrag[2], bfrag[2];
        for (int mt = 0; mt < 2; ++mt) {
            int m = waveM * 32 + mt * 16 + l15;
            v8bf lo = *(const v8bf*)(Asm + m * 32 + half * 8);
            v8bf hi = *(const v8bf*)(Asm + m * 32 + 16 + half * 8);
            afrag[mt] = cat8(lo, hi);
        }
        // B fragment: lane col n, K = {ko..ko+7, 16+ko..16+ko+7}, ko = half*8
        for (int nt = 0; nt < 2; ++nt) {
            int n = waveN * 32 + nt * 16 + l15;
            v8bf lo = *(const v8bf*)(Bsm + n * 32 + half * 8);
            v8bf hi = *(const v8bf*)(Bsm + n * 32 + 16 + half * 8);
            bfrag[nt] = cat8(lo, hi);
        }
        for (int mt = 0; mt < 2; ++mt)
            for (int nt = 0; nt < 2; ++nt)
                acc[mt][nt] = wmma_bf16(afrag[mt], bfrag[nt], acc[mt][nt]);
        __syncthreads();
    }

    // Epilogue: C element j <-> row (j + 8*half), col l15
    for (int mt = 0; mt < 2; ++mt)
        for (int nt = 0; nt < 2; ++nt) {
            int n = nBase + waveN * 32 + nt * 16 + l15;
            bf16_t* dst = out0;
            int n2 = n;
            if (kvMode && n >= DD) { dst = out1; n2 = n - DD; }
            int hh = n2 >> 6, d = n2 & 63;
            for (int j = 0; j < 8; ++j) {
                int m = mBase + waveM * 32 + mt * 16 + half * 8 + j;
                int b = m >> 10, s = m & 1023;
                float v = acc[mt][nt][j] * scale;
                dst[((size_t)(b * HH + hh) * SS + s) * DHH + d] = (bf16_t)v;
            }
        }
}

// ---------------------------------------------------------------------------
// Kernel 2: flash-attention over [B,H]; block = 4 waves x 16 query rows.
// Q pre-scaled by 1/sqrt(D). K tile goes global->LDS via ASYNC copy (natural
// [key][dh] layout == B layout for Q*K^T); V tile is transposed to [dh][key]
// (= B layout for P*V). P goes through per-wave LDS (C-layout -> A-layout).
// ---------------------------------------------------------------------------
__global__ __launch_bounds__(128)
void attn_kernel(const bf16_t* __restrict__ Qp, const bf16_t* __restrict__ Kp,
                 const bf16_t* __restrict__ Vp,
                 const int* __restrict__ l_en, const int* __restrict__ l_fr,
                 float* __restrict__ out)     // [B,S,D]
{
    __shared__ bf16_t Ksm[64 * 64];          // [key][dh]
    __shared__ bf16_t Vsm[64 * 64];          // [dh][key]
    __shared__ bf16_t Psm[4][16 * 64];       // per-wave P scratch

    const int tid  = threadIdx.x;
    const int lane = tid & 31;
    const int wave = tid >> 5;
    const int half = lane >> 4;
    const int l15  = lane & 15;

    const int qTile = blockIdx.x;            // 0..15
    const int h     = blockIdx.y;            // 0..15
    const int b     = blockIdx.z;            // 0..7

    const int Len = l_en[b];
    const int Lfr = l_fr[b];
    const int qBase = qTile * 64 + wave * 16;

    const bf16_t* Qhb = Qp + (size_t)(b * HH + h) * SS * DHH;
    const bf16_t* Khb = Kp + (size_t)(b * HH + h) * SS * DHH;
    const bf16_t* Vhb = Vp + (size_t)(b * HH + h) * SS * DHH;

    // Q A-fragments: rows qBase..+15, dh chunks {0..31},{32..63}
    v16bf qf[2];
    {
        int qm = qBase + l15;
        for (int c = 0; c < 2; ++c) {
            v8bf lo = *(const v8bf*)(Qhb + (size_t)qm * DHH + c * 32 + half * 8);
            v8bf hi = *(const v8bf*)(Qhb + (size_t)qm * DHH + c * 32 + 16 + half * 8);
            qf[c] = cat8(lo, hi);
        }
    }

    v8f o[4];
    for (int t = 0; t < 4; ++t) o[t] = splat8(0.f);
    float mrow[8], lrow[8];
    for (int j = 0; j < 8; ++j) { mrow[j] = -__builtin_inff(); lrow[j] = 0.f; }

    for (int kb = 0; kb < Len; kb += 64) {
        // Cooperative K/V tile load.
        // K: async global->LDS DMA (layout preserving, 16B per lane).
        // V: through VGPRs with element transpose into [dh][key].
        for (int i = 0; i < 4; ++i) {
            int c   = tid + i * 128;          // 0..511
            int key = c >> 3;                 // 0..63
            int d0  = (c & 7) * 8;            // 0..56
#if HAS_ASYNC_LDS
            __builtin_amdgcn_global_load_async_to_lds_b128(
                (AS1 v4i*)(Khb + (size_t)(kb + key) * DHH + d0),
                (AS3 v4i*)(Ksm + key * 64 + d0),
                0, 0);
#else
            v8bf k8 = *(const v8bf*)(Khb + (size_t)(kb + key) * DHH + d0);
            *(v8bf*)(Ksm + key * 64 + d0) = k8;
#endif
            v8bf v8 = *(const v8bf*)(Vhb + (size_t)(kb + key) * DHH + d0);
            for (int e = 0; e < 8; ++e) Vsm[(d0 + e) * 64 + key] = v8[e];
        }
#if HAS_ASYNC_LDS
        __builtin_amdgcn_s_wait_asynccnt(0);   // own async copies done
#endif
        __syncthreads();                        // everyone's copies done

        // Scores 16x64: four 16x16 C tiles, K-dim = DH in 2 chunks
        v8f sc[4];
        for (int t = 0; t < 4; ++t) {
            v8f a = splat8(0.f);
            for (int c = 0; c < 2; ++c) {
                v8bf lo = *(const v8bf*)(Ksm + (t * 16 + l15) * 64 + c * 32 + half * 8);
                v8bf hi = *(const v8bf*)(Ksm + (t * 16 + l15) * 64 + c * 32 + 16 + half * 8);
                a = wmma_bf16(qf[c], cat8(lo, hi), a);
            }
            int key = kb + t * 16 + l15;
            if (key >= Len) a = splat8(-__builtin_inff());   // key mask
            sc[t] = a;
        }

        // Online softmax; row j+8*half lives in this lane's half
        float mnew[8], rscale[8], rsum[8];
        for (int j = 0; j < 8; ++j) {
            float mx = fmaxf(fmaxf(sc[0][j], sc[1][j]), fmaxf(sc[2][j], sc[3][j]));
            for (int off = 1; off < 16; off <<= 1)
                mx = fmaxf(mx, __shfl_xor(mx, off, 32));
            mnew[j]   = fmaxf(mrow[j], mx);
            rscale[j] = __expf(mrow[j] - mnew[j]);
            rsum[j]   = 0.f;
        }
        for (int t = 0; t < 4; ++t)
            for (int j = 0; j < 8; ++j) {
                float p = __expf(sc[t][j] - mnew[j]);
                rsum[j] += p;
                Psm[wave][(j + half * 8) * 64 + t * 16 + l15] = (bf16_t)p;
            }
        for (int j = 0; j < 8; ++j) {
            float s = rsum[j];
            for (int off = 1; off < 16; off <<= 1)
                s += __shfl_xor(s, off, 32);
            lrow[j] = lrow[j] * rscale[j] + s;
            mrow[j] = mnew[j];
        }
        for (int t = 0; t < 4; ++t)
            for (int j = 0; j < 8; ++j)
                o[t][j] *= rscale[j];

        // P x V: A from Psm (A-layout), B from transposed Vsm
        v16bf pf[2];
        for (int c = 0; c < 2; ++c) {
            v8bf lo = *(const v8bf*)(&Psm[wave][l15 * 64 + c * 32 + half * 8]);
            v8bf hi = *(const v8bf*)(&Psm[wave][l15 * 64 + c * 32 + 16 + half * 8]);
            pf[c] = cat8(lo, hi);
        }
        for (int t = 0; t < 4; ++t)
            for (int c = 0; c < 2; ++c) {
                v8bf lo = *(const v8bf*)(Vsm + (t * 16 + l15) * 64 + c * 32 + half * 8);
                v8bf hi = *(const v8bf*)(Vsm + (t * 16 + l15) * 64 + c * 32 + 16 + half * 8);
                o[t] = wmma_bf16(pf[c], cat8(lo, hi), o[t]);
            }
        __syncthreads();
    }

    // Epilogue: normalize, zero invalid queries, write [B,S,H*DH] f32
    for (int j = 0; j < 8; ++j) {
        int q = qBase + half * 8 + j;
        float inv = (q < Lfr && lrow[j] > 0.f) ? (1.0f / lrow[j]) : 0.0f;
        for (int t = 0; t < 4; ++t) {
            int d = t * 16 + l15;
            out[((size_t)(b * SS + q)) * DD + h * DHH + d] = o[t][j] * inv;
        }
    }
}

// ---------------------------------------------------------------------------
extern "C" void kernel_launch(void* const* d_in, const int* in_sizes, int n_in,
                              void* d_out, int out_size, void* d_ws, size_t ws_size,
                              hipStream_t stream) {
    const float* en   = (const float*)d_in[0];
    const float* fr   = (const float*)d_in[1];
    const int*   l_en = (const int*)d_in[2];
    const int*   l_fr = (const int*)d_in[3];
    const float* WQw  = (const float*)d_in[4];
    const float* WQb  = (const float*)d_in[5];
    const float* WKVw = (const float*)d_in[6];
    const float* WKVb = (const float*)d_in[7];

    const size_t perTensor = (size_t)BB * HH * SS * DHH;   // 8.4M bf16 elems
    bf16_t* Qp = (bf16_t*)d_ws;
    bf16_t* Kp = Qp + perTensor;
    bf16_t* Vp = Kp + perTensor;

    const float scaleQ = 1.0f / 32.0f;   // 1/sqrt(D), folded into Q

    // Q = (fr . WQ^T + bQ) / sqrt(D)
    dim3 gq(BB * SS / 128, DD / 64);
    proj_gemm_kernel<<<gq, 256, 0, stream>>>(fr, WQw, WQb, Qp, nullptr, 0, scaleQ);
    // [K;V] = en . WKV^T + bKV
    dim3 gkv(BB * SS / 128, 2 * DD / 64);
    proj_gemm_kernel<<<gkv, 256, 0, stream>>>(en, WKVw, WKVb, Kp, Vp, 1, 1.0f);
    // attention
    dim3 ga(SS / 64, HH, BB);
    attn_kernel<<<ga, 128, 0, stream>>>(Qp, Kp, Vp, l_en, l_fr, (float*)d_out);
}
